// CausalSelfAttention_48137993453955
// MI455X (gfx1250) — matrix operand order repaired and checked by reading, not compiled, not run on hardware
//
#include <hip/hip_runtime.h>
#include <hip/hip_bf16.h>

// CausalSelfAttention for MI455X (gfx1250): f16 WMMA flash attention with
// TDM (tensor_load_to_lds) staging of K/V tiles into LDS.
// B=2, T=4096, C=512, H=8, Dh=64. Inputs/outputs f32; internal f16 WMMA,
// f32 accumulate. Workspace: Q,K,V head-major f16 (3x8MB) + y f16 (8MB).

#define TSEQ 4096
#define CDIM 512
#define NH   8
#define DH   64
#define NB   2
#define ROWS (NB * TSEQ)   // 8192
#define QKVC (3 * CDIM)    // 1536

typedef __attribute__((ext_vector_type(16))) _Float16 v16h;
typedef __attribute__((ext_vector_type(8)))  _Float16 v8h;
typedef __attribute__((ext_vector_type(8)))  float    v8f;
typedef __attribute__((ext_vector_type(4)))  float    v4f;

typedef unsigned int u32x4 __attribute__((ext_vector_type(4)));
typedef int          i32x4 __attribute__((ext_vector_type(4)));
typedef int          i32x8 __attribute__((ext_vector_type(8)));

#if defined(__has_builtin)
#if __has_builtin(__builtin_amdgcn_tensor_load_to_lds) && \
    __has_builtin(__builtin_amdgcn_s_wait_tensorcnt)
#define HAVE_TDM 1
#endif
#endif
#ifndef HAVE_TDM
#define HAVE_TDM 0
#endif

static __device__ __forceinline__ v8f wmma16(v16h a, v16h b, v8f c) {
  // (neg_a, A, neg_b, B, c_mod, C, reuse_a, reuse_b)
  return __builtin_amdgcn_wmma_f32_16x16x32_f16(false, a, false, b, (short)0, c,
                                                false, false);
}

// pack 4x v4f (f32) -> v16h (f16), in order
static __device__ __forceinline__ v16h cvt4(v4f a, v4f b, v4f c, v4f d) {
  v16h r;
#pragma unroll
  for (int i = 0; i < 4; ++i) {
    r[i]      = (_Float16)a[i];
    r[4 + i]  = (_Float16)b[i];
    r[8 + i]  = (_Float16)c[i];
    r[12 + i] = (_Float16)d[i];
  }
  return r;
}

static __device__ __forceinline__ v16h joinv(v8h lo, v8h hi) {
  v16h r;
#pragma unroll
  for (int i = 0; i < 8; ++i) { r[i] = lo[i]; r[8 + i] = hi[i]; }
  return r;
}

#if HAVE_TDM
// Issue a TDM 2D tile load (f16 elements): rows x cols tile, row stride
// `stride0` elements, from global `gsrc` into LDS byte offset `lds_off`.
// D# built per CDNA5 ISA 8.3/8.4 (group0: count/lds/global/type=2;
// group1: data_size=2B, tensor_dim0/1, tile_dim0/1, dim0 stride).
// This toolchain's builtin is the 6-arg form:
//   (u32x4 g0, i32x8 g1, i32x4 g2, i32x4 g3, i32x8 extra, i32 cpol)
static __device__ __forceinline__ void tdm_load_tile_f16(
    unsigned lds_off, const _Float16* gsrc, unsigned cols, unsigned rows,
    unsigned stride0, unsigned tensor_rows) {
  unsigned long long ga = (unsigned long long)gsrc;
  u32x4 g0;
  g0[0] = 1u;                                                  // count=1
  g0[1] = lds_off;                                             // lds_addr
  g0[2] = (unsigned)ga;                                        // ga[31:0]
  g0[3] = (unsigned)((ga >> 32) & 0x01FFFFFFu) | (2u << 30);   // ga[56:32]|type=2
  i32x8 g1;
  g1[0] = (int)(1u << 16);                                     // data_size=2B
  g1[1] = (int)((cols & 0xFFFFu) << 16);                       // tdim0[15:0]
  g1[2] = (int)(((cols >> 16) & 0xFFFFu) |
                ((tensor_rows & 0xFFFFu) << 16));              // tdim0hi|tdim1lo
  g1[3] = (int)(((tensor_rows >> 16) & 0xFFFFu) |
                ((cols & 0xFFFFu) << 16));                     // tdim1hi|tile0
  g1[4] = (int)(rows & 0xFFFFu);                               // tile1, tile2=0
  g1[5] = (int)stride0;                                        // dim0 stride
  g1[6] = 0;
  g1[7] = 0;
  i32x4 z4 = {0, 0, 0, 0};
  i32x8 z8 = {0, 0, 0, 0, 0, 0, 0, 0};
  __builtin_amdgcn_tensor_load_to_lds(g0, g1, z4, z4, z8, 0);
}
#endif

// ---------------------------------------------------------------------------
// Kernel 1: QKV projection. qkv[row,col] = x[row,:] . w_qkv[col,:] + b.
// One wave per 16x16 tile; scatter into head-major f16 Q/K/V.
// ---------------------------------------------------------------------------
__global__ __launch_bounds__(256) void qkv_proj_kernel(
    const float* __restrict__ x, const float* __restrict__ wqkv,
    const float* __restrict__ bqkv,
    _Float16* __restrict__ Qh, _Float16* __restrict__ Kh,
    _Float16* __restrict__ Vh) {
  const int lane = threadIdx.x & 31;
  const int wave = threadIdx.x >> 5;
  const int hi = lane >> 4;
  const int ln = lane & 15;
  const int rowBase = blockIdx.x * 16;
  const int colBase = (blockIdx.y * 8 + wave) * 16;

  const float* arow = x + (size_t)(rowBase + ln) * CDIM;
  const float* brow = wqkv + (size_t)(colBase + ln) * CDIM;

  v8f acc = {};
#pragma unroll 4
  for (int k = 0; k < CDIM; k += 32) {
    const float* ap = arow + k + 16 * hi;
    v16h a = cvt4(*(const v4f*)(ap),      *(const v4f*)(ap + 4),
                  *(const v4f*)(ap + 8),  *(const v4f*)(ap + 12));
    const float* bp = brow + k + 16 * hi;
    v16h b = cvt4(*(const v4f*)(bp),     *(const v4f*)(bp + 4),
                  *(const v4f*)(bp + 8), *(const v4f*)(bp + 12));
    acc = wmma16(a, b, acc);
  }

  const int s = colBase / CDIM;
  const int head = (colBase % CDIM) / DH;
  const int dBase = colBase % DH;
  _Float16* dst = (s == 0) ? Qh : (s == 1) ? Kh : Vh;
  const float bias = bqkv[colBase + ln];
#pragma unroll
  for (int r = 0; r < 8; ++r) {
    const int m = r + hi * 8;
    const int row = rowBase + m;
    const int bb = row >> 12;
    const int t = row & (TSEQ - 1);
    dst[(((size_t)bb * NH + head) * TSEQ + t) * DH + dBase + ln] =
        (_Float16)(acc[r] + bias);
  }
}

// ---------------------------------------------------------------------------
// Kernel 2: causal flash attention. 128 threads = 4 waves per block; block
// owns a 64-row q block, wave w owns rows [qBlock+16w, +16). K/V tiles
// (32x64 f16, 4KB each) staged into LDS by the Tensor Data Mover (wave 0
// issues tensor_load_to_lds; s_wait_tensorcnt 0 + barrier publishes them).
// Per tile: S = Q.K^T (4 wmma), causal mask, online softmax (shfl_xor row
// reductions in the 16-lane half-groups of the D layout), P relaid via a
// per-wave LDS scratch into an A fragment, O += P.V (4 wmma).
// ---------------------------------------------------------------------------
__global__ __launch_bounds__(128) void attn_kernel(
    const _Float16* __restrict__ Qh, const _Float16* __restrict__ Kh,
    const _Float16* __restrict__ Vh, _Float16* __restrict__ Yh) {
  // [0,4096)   : K tile  32x64 f16
  // [4096,8192): V tile  32x64 f16
  // [8192,...) : per-wave P scratch, 4 x 16x32 f16
  __shared__ __align__(128) _Float16 smem[2048 + 2048 + 4 * 512];
  _Float16 (*Kt)[DH] = (_Float16(*)[DH])smem;
  _Float16 (*Vt)[DH] = (_Float16(*)[DH])(smem + 2048);

  const int tid = threadIdx.x;
  const int lane = tid & 31;
  const int wave = tid >> 5;
  const int hi = lane >> 4;
  const int ln = lane & 15;
  _Float16* Pw = smem + 4096 + wave * 512;  // [16][32]

  const int bh = blockIdx.x;                // b*H + head
  const int qBlock = blockIdx.y * 64;
  const int qBase = qBlock + wave * 16;
  const size_t hoff = (size_t)bh * TSEQ * DH;
  const _Float16* Qb = Qh + hoff;
  const _Float16* Kb = Kh + hoff;
  const _Float16* Vb = Vh + hoff;

  // Q A-fragments (Dh=64 -> two 32-wide k-steps), held in registers.
  const _Float16* qrow = Qb + (size_t)(qBase + ln) * DH;
  v16h qa[2];
#pragma unroll
  for (int s = 0; s < 2; ++s) {
    const _Float16* qp = qrow + 32 * s + 16 * hi;
    qa[s] = joinv(*(const v8h*)(qp), *(const v8h*)(qp + 8));
  }

  float mstat[8], lstat[8];
#pragma unroll
  for (int r = 0; r < 8; ++r) { mstat[r] = -1e30f; lstat[r] = 0.f; }
  v8f o[4] = {{}, {}, {}, {}};
  const float scale = 0.125f;  // 1/sqrt(64)

  const int kvEnd = qBlock + 64;  // uniform bound for all 4 waves (mask covers)
  for (int kv = 0; kv < kvEnd; kv += 32) {
    __syncthreads();  // previous iteration's K/V tile reads are complete
#if HAVE_TDM
    if (wave == 0) {  // EXEC is ignored by TDM; one wave issues both tiles
      tdm_load_tile_f16(0u,    Kb + (size_t)kv * DH, DH, 32, DH, TSEQ - kv);
      tdm_load_tile_f16(4096u, Vb + (size_t)kv * DH, DH, 32, DH, TSEQ - kv);
      __builtin_amdgcn_s_wait_tensorcnt(0);
    }
#else
    // cooperative fallback: 256 v8h chunks per tile, 128 threads x 2
#pragma unroll
    for (int cc = 0; cc < 2; ++cc) {
      const int idx = tid + cc * 128;
      const int row = idx >> 3, col = (idx & 7) * 8;
      *(v8h*)&Kt[row][col] = *(const v8h*)(Kb + (size_t)(kv + row) * DH + col);
      *(v8h*)&Vt[row][col] = *(const v8h*)(Vb + (size_t)(kv + row) * DH + col);
    }
#endif
    __syncthreads();  // K/V tiles visible to all waves

    // S tiles from LDS K: columns [kv,kv+16) and [kv+16,kv+32)
    v8f s0 = {}, s1 = {};
#pragma unroll
    for (int s = 0; s < 2; ++s) {
      const _Float16* kp0 = &Kt[ln][32 * s + 16 * hi];
      const _Float16* kp1 = &Kt[16 + ln][32 * s + 16 * hi];
      v16h kb0 = joinv(*(const v8h*)kp0, *(const v8h*)(kp0 + 8));
      v16h kb1 = joinv(*(const v8h*)kp1, *(const v8h*)(kp1 + 8));
      s0 = wmma16(qa[s], kb0, s0);
      s1 = wmma16(qa[s], kb1, s1);
    }

#pragma unroll
    for (int r = 0; r < 8; ++r) {
      const int m = r + hi * 8;
      const int q = qBase + m;
      float e0 = (kv + ln <= q)      ? s0[r] * scale : -1e30f;
      float e1 = (kv + 16 + ln <= q) ? s1[r] * scale : -1e30f;
      float rm = fmaxf(e0, e1);
#pragma unroll
      for (int msk = 1; msk < 16; msk <<= 1)
        rm = fmaxf(rm, __shfl_xor(rm, msk, 32));
      const float mnew = fmaxf(mstat[r], rm);
      const float corr = __expf(mstat[r] - mnew);
      const float p0 = __expf(e0 - mnew);
      const float p1 = __expf(e1 - mnew);
      float rs = p0 + p1;
#pragma unroll
      for (int msk = 1; msk < 16; msk <<= 1)
        rs += __shfl_xor(rs, msk, 32);
      lstat[r] = lstat[r] * corr + rs;
      mstat[r] = mnew;
      o[0][r] *= corr; o[1][r] *= corr; o[2][r] *= corr; o[3][r] *= corr;
      Pw[m * 32 + ln] = (_Float16)p0;
      Pw[m * 32 + 16 + ln] = (_Float16)p1;
    }
    __syncthreads();  // publish P (cross-lane) and order vs. Vt reads

    // P as A-fragment (M=ln, K=h+16*hi over the 32 kv cols)
    const _Float16* pr = &Pw[ln * 32 + 16 * hi];
    v16h pf = joinv(*(const v8h*)pr, *(const v8h*)(pr + 8));

#pragma unroll
    for (int t = 0; t < 4; ++t) {  // output columns dBase = 16*t
      v16h vf;
#pragma unroll
      for (int h = 0; h < 16; ++h)  // B-fragment: K=h+16*hi, N=ln (from LDS)
        vf[h] = Vt[h + 16 * hi][t * 16 + ln];
      o[t] = wmma16(pf, vf, o[t]);
    }
  }

  // normalize, merge heads into y[B,T,C] (f16)
  const int b = bh >> 3;
  const int head = bh & 7;
#pragma unroll
  for (int r = 0; r < 8; ++r) {
    const int m = r + hi * 8;
    const float inv = 1.f / lstat[r];
    _Float16* yr = Yh + (size_t)(b * TSEQ + qBase + m) * CDIM + head * DH;
#pragma unroll
    for (int t = 0; t < 4; ++t)
      yr[t * 16 + ln] = (_Float16)(o[t][r] * inv);
  }
}

// ---------------------------------------------------------------------------
// Kernel 3: output projection. out = y(f16) . w_out^T + b_out, f32 result.
// ---------------------------------------------------------------------------
__global__ __launch_bounds__(256) void out_proj_kernel(
    const _Float16* __restrict__ Yh, const float* __restrict__ wout,
    const float* __restrict__ bout, float* __restrict__ out) {
  const int lane = threadIdx.x & 31;
  const int wave = threadIdx.x >> 5;
  const int hi = lane >> 4;
  const int ln = lane & 15;
  const int rowBase = blockIdx.x * 16;
  const int colBase = (blockIdx.y * 8 + wave) * 16;

  const _Float16* arow = Yh + (size_t)(rowBase + ln) * CDIM;
  const float* brow = wout + (size_t)(colBase + ln) * CDIM;

  v8f acc = {};
#pragma unroll 4
  for (int k = 0; k < CDIM; k += 32) {
    const _Float16* ap = arow + k + 16 * hi;
    v16h a = joinv(*(const v8h*)ap, *(const v8h*)(ap + 8));
    const float* bp = brow + k + 16 * hi;
    v16h b = cvt4(*(const v4f*)bp,       *(const v4f*)(bp + 4),
                  *(const v4f*)(bp + 8), *(const v4f*)(bp + 12));
    acc = wmma16(a, b, acc);
  }
  const float bias = bout[colBase + ln];
#pragma unroll
  for (int r = 0; r < 8; ++r) {
    const int m = r + hi * 8;
    out[(size_t)(rowBase + m) * CDIM + colBase + ln] = acc[r] + bias;
  }
}

extern "C" void kernel_launch(void* const* d_in, const int* in_sizes, int n_in,
                              void* d_out, int out_size, void* d_ws,
                              size_t ws_size, hipStream_t stream) {
  (void)in_sizes; (void)n_in; (void)out_size; (void)ws_size;
  const float* x    = (const float*)d_in[0];
  const float* wqkv = (const float*)d_in[1];
  const float* bqkv = (const float*)d_in[2];
  const float* wout = (const float*)d_in[3];
  const float* bout = (const float*)d_in[4];
  float* out = (float*)d_out;

  // workspace layout (f16): Q | K | V | y  = 4 * 8MB = 32MB
  const size_t headElems = (size_t)NB * NH * TSEQ * DH;  // 4,194,304
  _Float16* Qh = (_Float16*)d_ws;
  _Float16* Kh = Qh + headElems;
  _Float16* Vh = Kh + headElems;
  _Float16* Yh = Vh + headElems;

  dim3 g1(ROWS / 16, QKVC / (16 * 8));  // (512, 12), 8 waves/block
  qkv_proj_kernel<<<g1, 256, 0, stream>>>(x, wqkv, bqkv, Qh, Kh, Vh);

  dim3 g2(NB * NH, TSEQ / 64);          // (16, 64), 4 waves/block
  attn_kernel<<<g2, 128, 0, stream>>>(Qh, Kh, Vh, Yh);

  dim3 g3(ROWS / 16, CDIM / (16 * 8));  // (512, 4), 8 waves/block
  out_proj_kernel<<<g3, 256, 0, stream>>>(Yh, wout, bout, out);
}
